// SymbolicTransformer_38543036514921
// MI455X (gfx1250) — matrix operand order repaired and checked by reading, not compiled, not run on hardware
//
#include <hip/hip_runtime.h>
#include <hip/hip_bf16.h>

typedef __attribute__((ext_vector_type(16))) _Float16 v16h;
typedef __attribute__((ext_vector_type(8)))  _Float16 v8h;
typedef __attribute__((ext_vector_type(8)))  float    v8f;

static constexpr int S_SEQ = 10;       // sequence length
static constexpr int NN    = 20000;    // nodes per slice
static constexpr int NE    = 320000;   // edges per slice
static constexpr int D     = 64;       // hidden dim
static constexpr int DFF   = 2048;     // feedforward dim
static constexpr int NG    = 64;       // graphs per slice (= transformer batch)
static constexpr int T     = S_SEQ * NG; // 640 tokens
static constexpr int NHEAD = 4;
static constexpr int HD    = D / NHEAD;  // 16

static __device__ __forceinline__ void atomic_add_f32(float* p, float v) {
  __hip_atomic_fetch_add(p, v, __ATOMIC_RELAXED, __HIP_MEMORY_SCOPE_AGENT);
}

// ---------------- elementwise / init kernels ----------------

__global__ void k_fill(float* p, float v, int n) {
  int i = blockIdx.x * blockDim.x + threadIdx.x;
  if (i < n) p[i] = v;
}

__global__ void k_fill_bias(float* p, const float* __restrict__ b, int rows) {
  // p[rows][D] = broadcast bias[D]
  int i = blockIdx.x * blockDim.x + threadIdx.x;
  if (i < rows * D) p[i] = b[i & (D - 1)];
}

__global__ void k_cvt16(_Float16* dst, const float* __restrict__ src, int n) {
  int i = blockIdx.x * blockDim.x + threadIdx.x;
  if (i < n) dst[i] = (_Float16)src[i];
}

__global__ void k_relu_cvt16(_Float16* dst, const float* __restrict__ src, int n) {
  int i = blockIdx.x * blockDim.x + threadIdx.x;
  if (i < n) dst[i] = (_Float16)fmaxf(src[i], 0.f);
}

__global__ void k_transpose_cvt16(_Float16* dst, const float* __restrict__ src,
                                  int rows, int cols) {
  // dst[c][r] = src[r][c]
  int i = blockIdx.x * blockDim.x + threadIdx.x;
  if (i < rows * cols) {
    int r = i / cols, c = i - r * cols;
    dst[(size_t)c * rows + r] = (_Float16)src[i];
  }
}

// ---------------- GCN kernels ----------------

__global__ void k_deg(const int* __restrict__ dst, float* deg, int E) {
  int i = blockIdx.x * blockDim.x + threadIdx.x;
  if (i < E) atomic_add_f32(&deg[dst[i]], 1.f);
}

__global__ void k_dinv(float* deg, int n) {
  int i = blockIdx.x * blockDim.x + threadIdx.x;
  if (i < n) { float d = deg[i]; deg[i] = (d > 0.f) ? rsqrtf(d) : 0.f; }
}

__global__ void k_gcn1_lin(const float* __restrict__ x, const float* __restrict__ w,
                           float* __restrict__ h, int n) {
  // x[n][2] @ w[2][64]
  int i = blockIdx.x * blockDim.x + threadIdx.x;
  if (i < n * D) {
    int node = i >> 6, f = i & 63;
    h[i] = x[2 * node] * w[f] + x[2 * node + 1] * w[D + f];
  }
}

__global__ void k_scatter(const float* __restrict__ h, const float* __restrict__ dinv,
                          const int* __restrict__ src, const int* __restrict__ dst,
                          float* acc, int E, int n) {
  // msg = h[src] * dinv[src]*dinv[dst], scatter-add to dst; edges E..E+n-1 are self loops
  const int per = D / 4;
  int i = blockIdx.x * blockDim.x + threadIdx.x;
  if (i >= (E + n) * per) return;
  int e = i / per, f4 = (i - e * per) * 4;
  int sI, dI;
  if (e < E) { sI = src[e]; dI = dst[e]; } else { sI = dI = e - E; }
  float c = dinv[sI] * dinv[dI];
  const float4 hv = *(const float4*)(h + (size_t)sI * D + f4);
  float* o = acc + (size_t)dI * D + f4;
  atomic_add_f32(o + 0, hv.x * c);
  atomic_add_f32(o + 1, hv.y * c);
  atomic_add_f32(o + 2, hv.z * c);
  atomic_add_f32(o + 3, hv.w * c);
}

__global__ void k_pool(const float* __restrict__ acc, const int* __restrict__ batch,
                       float* pool, float* cnt, int n) {
  // relu + per-graph segment sum
  const int per = D / 4;
  int i = blockIdx.x * blockDim.x + threadIdx.x;
  if (i >= n * per) return;
  int node = i / per, f4 = (i - node * per) * 4;
  int g = batch[node];
  const float4 v = *(const float4*)(acc + (size_t)node * D + f4);
  float* o = pool + (size_t)g * D + f4;
  atomic_add_f32(o + 0, fmaxf(v.x, 0.f));
  atomic_add_f32(o + 1, fmaxf(v.y, 0.f));
  atomic_add_f32(o + 2, fmaxf(v.z, 0.f));
  atomic_add_f32(o + 3, fmaxf(v.w, 0.f));
  if (f4 == 0) atomic_add_f32(&cnt[g], 1.f);
}

__global__ void k_reps(const float* __restrict__ pool, const float* __restrict__ cnt,
                       const float* __restrict__ pos, float* xo) {
  int i = blockIdx.x * blockDim.x + threadIdx.x;
  if (i >= NG * D) return;
  int g = i >> 6, f = i & 63;
  float c = fmaxf(cnt[g], 1.f);
  xo[i] = pool[i] / c + pos[f];
}

// ---------------- WMMA GEMM: C[M][N] = A[M][K] * Bt[N][K]^T (+bias[N]) (+relu) ----------------
// A, Bt row-major f16; M,N multiples of 16; K multiple of 32. One wave per 16x16 C tile.

__global__ __launch_bounds__(256)
void k_gemm(const _Float16* __restrict__ A, const _Float16* __restrict__ Bt,
            const float* __restrict__ bias, float* __restrict__ C,
            int M, int N, int K, int relu) {
  const int lane = threadIdx.x & 31;
  const int wave = blockIdx.x * (blockDim.x >> 5) + (threadIdx.x >> 5);
  const int ntN  = N >> 4;
  const int mt   = wave / ntN;
  const int nt   = wave - mt * ntN;
  if (mt * 16 >= M) return;                  // wave-uniform guard (EXEC stays all-ones)
  const int rm   = lane & 15;                // row (A) / col (B) within tile
  const int half = lane >> 4;                // K-half selector per ISA lane layout
  const _Float16* ar = A  + (size_t)(mt * 16 + rm) * K;
  const _Float16* br = Bt + (size_t)(nt * 16 + rm) * K;
  v8f acc = {};
  for (int k0 = 0; k0 < K; k0 += 32) {
    union { v16h v; v8h h[2]; } a, b;
    // A 16x32 f16: lanes 0-15 hold K 0..7 (v0-3) & 16..23 (v4-7); lanes 16-31 offset +8
    a.h[0] = *(const v8h*)(ar + k0 + 8 * half);
    a.h[1] = *(const v8h*)(ar + k0 + 16 + 8 * half);
    // B 32x16 f16 (loaded from Bt[N][K]): same K layout as A (K 0..7 & 16..23; lanes 16-31 offset +8)
    b.h[0] = *(const v8h*)(br + k0 + 8 * half);
    b.h[1] = *(const v8h*)(br + k0 + 16 + 8 * half);
    acc = __builtin_amdgcn_wmma_f32_16x16x32_f16(false, a.v, false, b.v,
                                                 (short)0, acc, false, false);
  }
  const int col = nt * 16 + rm;
  const float bv = bias ? bias[col] : 0.f;
  float* cp = C + (size_t)(mt * 16 + 8 * half) * N + col;
#pragma unroll
  for (int r2 = 0; r2 < 8; ++r2) {
    float v = acc[r2] + bv;
    if (relu) v = fmaxf(v, 0.f);
    cp[(size_t)r2 * N] = v;
  }
}

// ---------------- transformer kernels ----------------

__global__ void k_attn(const float* __restrict__ qkv, float* __restrict__ o) {
  // qkv: [T][3*D] token-major (token = s*NG + b); one thread per (b, head, s)
  int tid = blockIdx.x * blockDim.x + threadIdx.x;
  if (tid >= NG * NHEAD * S_SEQ) return;
  int b = tid / (NHEAD * S_SEQ);
  int r = tid - b * (NHEAD * S_SEQ);
  int h = r / S_SEQ;
  int s = r - h * S_SEQ;
  const float scale = rsqrtf((float)HD);
  float q[HD];
  const float* qp = qkv + (size_t)(s * NG + b) * (3 * D) + h * HD;
#pragma unroll
  for (int j = 0; j < HD; ++j) q[j] = qp[j];
  float sc[S_SEQ];
  float mx = -3.4e38f;
#pragma unroll
  for (int t = 0; t < S_SEQ; ++t) {
    const float* kp = qkv + (size_t)(t * NG + b) * (3 * D) + D + h * HD;
    float dot = 0.f;
#pragma unroll
    for (int j = 0; j < HD; ++j) dot += q[j] * kp[j];
    dot *= scale;
    sc[t] = dot;
    mx = fmaxf(mx, dot);
  }
  float sum = 0.f;
#pragma unroll
  for (int t = 0; t < S_SEQ; ++t) { sc[t] = __expf(sc[t] - mx); sum += sc[t]; }
  float inv = 1.f / sum;
  float acc[HD];
#pragma unroll
  for (int j = 0; j < HD; ++j) acc[j] = 0.f;
#pragma unroll
  for (int t = 0; t < S_SEQ; ++t) {
    const float* vp = qkv + (size_t)(t * NG + b) * (3 * D) + 2 * D + h * HD;
    float w = sc[t] * inv;
#pragma unroll
    for (int j = 0; j < HD; ++j) acc[j] += w * vp[j];
  }
  float* op = o + (size_t)(s * NG + b) * D + h * HD;
#pragma unroll
  for (int j = 0; j < HD; ++j) op[j] = acc[j];
}

__global__ void k_add_ln(const float* __restrict__ x, const float* __restrict__ r,
                         const float* __restrict__ w, const float* __restrict__ b,
                         float* __restrict__ y) {
  int t = blockIdx.x * blockDim.x + threadIdx.x;
  if (t >= T) return;
  const float* xp = x + (size_t)t * D;
  const float* rp = r + (size_t)t * D;
  float m = 0.f;
  for (int f = 0; f < D; ++f) m += xp[f] + rp[f];
  m *= (1.f / D);
  float v = 0.f;
  for (int f = 0; f < D; ++f) { float d = xp[f] + rp[f] - m; v += d * d; }
  v *= (1.f / D);
  float is = rsqrtf(v + 1e-5f);
  float* yp = y + (size_t)t * D;
  for (int f = 0; f < D; ++f) yp[f] = (xp[f] + rp[f] - m) * is * w[f] + b[f];
}

__global__ void k_final(const float* __restrict__ x, const float* __restrict__ fw,
                        const float* __restrict__ fb, float* __restrict__ out) {
  int b = blockIdx.x * blockDim.x + threadIdx.x;
  if (b >= NG) return;
  const float* xp = x + (size_t)((S_SEQ - 1) * NG + b) * D;
  float acc = 0.f;
  for (int k = 0; k < D; ++k) acc += xp[k] * fw[k];
  out[b] = acc + fb[0];
}

// ---------------- host side ----------------

static inline void gemm(const _Float16* A, const _Float16* Bt, const float* bias,
                        float* C, int M, int N, int K, int relu, hipStream_t st) {
  int waves  = (M / 16) * (N / 16);
  int blocks = (waves + 7) / 8;   // 8 waves (256 threads) per block
  k_gemm<<<blocks, 256, 0, st>>>(A, Bt, bias, C, M, N, K, relu);
}

extern "C" void kernel_launch(void* const* d_in, const int* in_sizes, int n_in,
                              void* d_out, int out_size, void* d_ws, size_t ws_size,
                              hipStream_t stream) {
  (void)in_sizes; (void)n_in; (void)out_size; (void)ws_size;
  const float* x    = (const float*)d_in[0];
  const int*   ei   = (const int*)d_in[1];
  const int*   bat  = (const int*)d_in[2];
  const float* g1w  = (const float*)d_in[3];
  const float* g1b  = (const float*)d_in[4];
  const float* g2w  = (const float*)d_in[5];
  const float* g2b  = (const float*)d_in[6];
  const float* pos  = (const float*)d_in[7];
  const float* qkvw = (const float*)d_in[8];
  const float* qkvb = (const float*)d_in[9];
  const float* aow  = (const float*)d_in[10];
  const float* aob  = (const float*)d_in[11];
  const float* l1w  = (const float*)d_in[12];
  const float* l1b  = (const float*)d_in[13];
  const float* l2w  = (const float*)d_in[14];
  const float* l2b  = (const float*)d_in[15];
  const float* f1w  = (const float*)d_in[16];
  const float* f1b  = (const float*)d_in[17];
  const float* f2w  = (const float*)d_in[18];
  const float* f2b  = (const float*)d_in[19];
  const float* fcw  = (const float*)d_in[20];
  const float* fcb  = (const float*)d_in[21];
  float* out = (float*)d_out;

  char* ws = (char*)d_ws;
  size_t off = 0;
  auto alloc = [&](size_t bytes) -> char* {
    char* p = ws + off;
    off = (off + bytes + 255) & ~size_t(255);
    return p;
  };
  float*    deg    = (float*)alloc((size_t)NN * 4);
  float*    bufA   = (float*)alloc((size_t)NN * D * 4);  // gcn linear-out / h2lin
  float*    bufB   = (float*)alloc((size_t)NN * D * 4);  // scatter accumulator
  _Float16* h1h    = (_Float16*)alloc((size_t)NN * D * 2);
  _Float16* w2t    = (_Float16*)alloc((size_t)D * D * 2);
  _Float16* qkvwh  = (_Float16*)alloc((size_t)2 * 3 * D * D * 2);
  _Float16* owh    = (_Float16*)alloc((size_t)2 * D * D * 2);
  _Float16* f1wh   = (_Float16*)alloc((size_t)2 * DFF * D * 2);
  _Float16* f2wh   = (_Float16*)alloc((size_t)2 * D * DFF * 2);
  float*    xbuf   = (float*)alloc((size_t)T * D * 4);
  float*    x2buf  = (float*)alloc((size_t)T * D * 4);
  _Float16* xh     = (_Float16*)alloc((size_t)T * D * 2);
  float*    qkvbuf = (float*)alloc((size_t)T * 3 * D * 4);
  float*    attno  = (float*)alloc((size_t)T * D * 4);
  float*    oproj  = (float*)alloc((size_t)T * D * 4);
  float*    ffbuf  = (float*)alloc((size_t)T * DFF * 4);
  _Float16* ffh    = (_Float16*)alloc((size_t)T * DFF * 2);
  float*    ff2buf = (float*)alloc((size_t)T * D * 4);
  float*    pool   = (float*)alloc((size_t)NG * D * 4);
  float*    cnt    = (float*)alloc((size_t)NG * 4);

  const int TB = 256;
  auto nb = [&](long n) { return (int)((n + TB - 1) / TB); };

  // one-time weight prep (f16, Bt layout = [out][in])
  k_transpose_cvt16<<<nb(D * D), TB, 0, stream>>>(w2t, g2w, D, D);
  k_cvt16<<<nb(2 * 3 * D * D), TB, 0, stream>>>(qkvwh, qkvw, 2 * 3 * D * D);
  k_cvt16<<<nb(2 * D * D), TB, 0, stream>>>(owh, aow, 2 * D * D);
  k_cvt16<<<nb(2 * DFF * D), TB, 0, stream>>>(f1wh, f1w, 2 * DFF * D);
  k_cvt16<<<nb(2 * D * DFF), TB, 0, stream>>>(f2wh, f2w, 2 * D * DFF);

  // ---- GCN encoder, one slice at a time (keeps workspace ~23 MB) ----
  for (int s = 0; s < S_SEQ; ++s) {
    const float* xs   = x + (size_t)s * NN * 2;
    const int*   srcI = ei + (size_t)s * 2 * NE;
    const int*   dstI = srcI + NE;
    const int*   bs   = bat + (size_t)s * NN;

    k_fill<<<nb(NN), TB, 0, stream>>>(deg, 1.f, NN);               // self-loop degree
    k_deg<<<nb(NE), TB, 0, stream>>>(dstI, deg, NE);
    k_dinv<<<nb(NN), TB, 0, stream>>>(deg, NN);                    // deg -> deg^-0.5

    // conv1: x@W1, normalized scatter + bias, relu -> f16
    k_gcn1_lin<<<nb((long)NN * D), TB, 0, stream>>>(xs, g1w, bufA, NN);
    k_fill_bias<<<nb((long)NN * D), TB, 0, stream>>>(bufB, g1b, NN);
    k_scatter<<<nb((long)(NE + NN) * (D / 4)), TB, 0, stream>>>(bufA, deg, srcI, dstI, bufB, NE, NN);
    k_relu_cvt16<<<nb((long)NN * D), TB, 0, stream>>>(h1h, bufB, NN * D);

    // conv2: WMMA GEMM h1@W2, normalized scatter + bias
    gemm(h1h, w2t, nullptr, bufA, NN, D, D, 0, stream);
    k_fill_bias<<<nb((long)NN * D), TB, 0, stream>>>(bufB, g2b, NN);
    k_scatter<<<nb((long)(NE + NN) * (D / 4)), TB, 0, stream>>>(bufA, deg, srcI, dstI, bufB, NE, NN);

    // relu + mean pool per graph, + positional embedding -> token buffer
    k_fill<<<nb(NG * D), TB, 0, stream>>>(pool, 0.f, NG * D);
    k_fill<<<nb(NG), TB, 0, stream>>>(cnt, 0.f, NG);
    k_pool<<<nb((long)NN * (D / 4)), TB, 0, stream>>>(bufB, bs, pool, cnt, NN);
    k_reps<<<nb(NG * D), TB, 0, stream>>>(pool, cnt, pos + s * D, xbuf + (size_t)s * NG * D);
  }

  // ---- 2 transformer encoder layers (all GEMMs via WMMA) ----
  for (int l = 0; l < 2; ++l) {
    k_cvt16<<<nb(T * D), TB, 0, stream>>>(xh, xbuf, T * D);
    gemm(xh, qkvwh + (size_t)l * 3 * D * D, qkvb + l * 3 * D, qkvbuf, T, 3 * D, D, 0, stream);
    k_attn<<<nb(NG * NHEAD * S_SEQ), TB, 0, stream>>>(qkvbuf, attno);
    k_cvt16<<<nb(T * D), TB, 0, stream>>>(xh, attno, T * D);
    gemm(xh, owh + (size_t)l * D * D, aob + l * D, oproj, T, D, D, 0, stream);
    k_add_ln<<<nb(T), TB, 0, stream>>>(xbuf, oproj, l1w + l * D, l1b + l * D, x2buf);
    k_cvt16<<<nb(T * D), TB, 0, stream>>>(xh, x2buf, T * D);
    gemm(xh, f1wh + (size_t)l * DFF * D, f1b + l * DFF, ffbuf, T, DFF, D, 1, stream);
    k_cvt16<<<nb((long)T * DFF), TB, 0, stream>>>(ffh, ffbuf, T * DFF);
    gemm(ffh, f2wh + (size_t)l * D * DFF, f2b + l * D, ff2buf, T, D, DFF, 0, stream);
    k_add_ln<<<nb(T), TB, 0, stream>>>(x2buf, ff2buf, l2w + l * D, l2b + l * D, xbuf);
  }

  k_final<<<nb(NG), TB, 0, stream>>>(xbuf, fcw, fcb, out);
}
